// MultiheadAttention_90237262889735
// MI455X (gfx1250) — matrix operand order repaired and checked by reading, not compiled, not run on hardware
//
#include <hip/hip_runtime.h>

#define S_LEN   2048
#define DM      1024
#define NH      16
#define HD      64
#define SM_SCALE 0.125f   // 1/sqrt(64)

typedef __bf16 bf16_t;
typedef __attribute__((ext_vector_type(16))) __bf16 v16bf;
typedef __attribute__((ext_vector_type(8)))  __bf16 v8bf;
typedef __attribute__((ext_vector_type(8)))  float  v8f;

__device__ __forceinline__ unsigned short f2bf_bits(float f) {
  unsigned u = __float_as_uint(f);
  u += 0x7FFFu + ((u >> 16) & 1u);          // round-to-nearest-even
  return (unsigned short)(u >> 16);
}
__device__ __forceinline__ bf16_t bits2bf(unsigned short s) {
  union { unsigned short u; bf16_t b; } x; x.u = s; return x.b;
}

__device__ __forceinline__ v8f wmma_bf16(v16bf a, v16bf b, v8f c) {
  return __builtin_amdgcn_wmma_f32_16x16x32_bf16(false, a, false, b, (short)0, c,
                                                 false, false);
}

// ---- A fragment: 16x32 (MxK) bf16.  lane<16: row M=lane, K elems
// {h*8..h*8+7} then {16+h*8..16+h*8+7}  (CDNA5 ISA 16-bit A layout)
__device__ __forceinline__ v16bf load_afrag(const bf16_t* A, int lda, int row0, int k0, int lane) {
  int r = lane & 15, h = lane >> 4;
  const bf16_t* p = A + (long)(row0 + r) * lda + k0 + h * 8;
  union { v16bf v; v8bf x[2]; } u;
  u.x[0] = *(const v8bf*)p;
  u.x[1] = *(const v8bf*)(p + 16);
  return u.v;
}
// ---- B fragment (NT): source row-major [N,K]; output column n = lane%16,
// K elems {h*8..h*8+7} then {16+h*8..16+h*8+7} (same order as A)
__device__ __forceinline__ v16bf load_bfrag(const bf16_t* Bs, int ldb, int col0, int k0, int lane) {
  int r = lane & 15, h = lane >> 4;
  const bf16_t* p = Bs + (long)(col0 + r) * ldb + k0 + h * 8;
  union { v16bf v; v8bf x[2]; } u;
  u.x[0] = *(const v8bf*)p;
  u.x[1] = *(const v8bf*)(p + 16);
  return u.v;
}

// ============ fp32 -> bf16 bulk down-convert (bandwidth bound, one pass) ============
__global__ __launch_bounds__(256) void cvt_bf16_kernel(const float* __restrict__ in,
                                                       bf16_t* __restrict__ out, int n) {
  int i = (blockIdx.x * 256 + threadIdx.x) * 8;
  if (i + 8 <= n) {
    v8f f = *(const v8f*)(in + i);
    union { v8bf v; unsigned short s[8]; } u;
#pragma unroll
    for (int j = 0; j < 8; ++j) u.s[j] = f2bf_bits(f[j]);
    *(v8bf*)(out + i) = u.v;
  }
}

// ============ C[M,N] = A[M,K] * W[N,K]^T + bias  (A, W bf16; acc fp32) ============
// EPI 0: scatter Q (scaled), K to [b,h,s,d] and V transposed to [b,h,d,s], bf16
// EPI 1: plain fp32 row-major output
template <int EPI>
__global__ __launch_bounds__(256) void gemm_nt_kernel(
    const bf16_t* __restrict__ A, const bf16_t* __restrict__ W, const float* __restrict__ bias,
    int M, int N, int K,
    bf16_t* __restrict__ qb, bf16_t* __restrict__ kb, bf16_t* __restrict__ vtb,
    float* __restrict__ outp) {
  int lane = threadIdx.x & 31;
  int wave = threadIdx.x >> 5;
  int wm = wave & 3, wn = wave >> 2;             // 4 waves along M, 2 along N
  int m0 = blockIdx.y * 128 + wm * 32;
  int n0 = blockIdx.x * 128 + wn * 64;
  v8f acc[2][4] = {};
  for (int k = 0; k < K; k += 32) {
    v16bf a0 = load_afrag(A, K, m0,      k, lane);
    v16bf a1 = load_afrag(A, K, m0 + 16, k, lane);
#pragma unroll
    for (int t = 0; t < 4; ++t) {
      v16bf b = load_bfrag(W, K, n0 + t * 16, k, lane);
      acc[0][t] = wmma_bf16(a0, b, acc[0][t]);
      acc[1][t] = wmma_bf16(a1, b, acc[1][t]);
    }
  }
  int r = lane & 15, h = lane >> 4;
#pragma unroll
  for (int mi = 0; mi < 2; ++mi) {
#pragma unroll
    for (int t = 0; t < 4; ++t) {
      int n = n0 + t * 16 + r;
      float bv = bias[n];
#pragma unroll
      for (int i = 0; i < 8; ++i) {
        int m = m0 + mi * 16 + i + 8 * h;        // C/D layout: half1 holds rows 8..15
        float v = acc[mi][t][i] + bv;
        if (EPI == 0) {
          int sec = n >> 10;                     // 0:Q 1:K 2:V
          int nn = n & 1023;
          int hh = nn >> 6, dd = nn & 63;
          int bb = m >> 11, s = m & 2047;
          long bh = (long)bb * NH + hh;
          if (sec == 0)      qb[(bh * S_LEN + s) * HD + dd] = bits2bf(f2bf_bits(v * SM_SCALE));
          else if (sec == 1) kb[(bh * S_LEN + s) * HD + dd] = bits2bf(f2bf_bits(v));
          else               vtb[(bh * HD + dd) * S_LEN + s] = bits2bf(f2bf_bits(v));
        } else {
          outp[(long)m * N + n] = v;
        }
      }
    }
  }
}

// ============ Flash attention: per wave = 16 query rows, KV blocks of 64 ============
__global__ __launch_bounds__(256) void attn_kernel(
    const bf16_t* __restrict__ Q, const bf16_t* __restrict__ Kc,
    const bf16_t* __restrict__ Vt, bf16_t* __restrict__ ctx) {
  __shared__ __align__(16) unsigned short Pb[8][16 * 64];   // per-wave P tile (bf16 bits)
  int lane = threadIdx.x & 31;
  int wave = threadIdx.x >> 5;
  int r = lane & 15, h = lane >> 4;
  int bh = blockIdx.y;                                   // b*16 + head
  int m0 = blockIdx.x * 128 + wave * 16;                 // query row base
  const bf16_t* q  = Q  + (long)bh * S_LEN * HD;
  const bf16_t* kk = Kc + (long)bh * S_LEN * HD;
  const bf16_t* vt = Vt + (long)bh * HD * S_LEN;
  // Q fragments resident for whole kernel (d = 0..31, 32..63)
  v16bf qa0 = load_afrag(q, HD, m0, 0,  lane);
  v16bf qa1 = load_afrag(q, HD, m0, 32, lane);
  v8f acc[4] = {};
  float mrow[8], lrow[8];
#pragma unroll
  for (int i = 0; i < 8; ++i) { mrow[i] = -1e30f; lrow[i] = 0.f; }
  unsigned short* P = &Pb[wave][0];
  for (int s0 = 0; s0 < S_LEN; s0 += 64) {
    // scores = (Q*scale) @ K^T  : 8 WMMAs
    v8f sc[4] = {};
#pragma unroll
    for (int t = 0; t < 4; ++t) {
      sc[t] = wmma_bf16(qa0, load_bfrag(kk, HD, s0 + t * 16, 0,  lane), sc[t]);
      sc[t] = wmma_bf16(qa1, load_bfrag(kk, HD, s0 + t * 16, 32, lane), sc[t]);
    }
    // online softmax; row i+8*h lives in VGPR i, reduced across the 16-lane half
#pragma unroll
    for (int i = 0; i < 8; ++i) {
      float mx = fmaxf(fmaxf(sc[0][i], sc[1][i]), fmaxf(sc[2][i], sc[3][i]));
#pragma unroll
      for (int d = 1; d < 16; d <<= 1) mx = fmaxf(mx, __shfl_xor(mx, d, 32));
      float mnew = fmaxf(mrow[i], mx);
      float sca  = __expf(mrow[i] - mnew);
      mrow[i] = mnew;
      int row = i + 8 * h;
      float rs = 0.f;
#pragma unroll
      for (int t = 0; t < 4; ++t) {
        float p = __expf(sc[t][i] - mnew);
        rs += p;
        P[row * 64 + t * 16 + r] = f2bf_bits(p);   // C-layout -> LDS row-major
      }
#pragma unroll
      for (int d = 1; d < 16; d <<= 1) rs += __shfl_xor(rs, d, 32);
      lrow[i] = lrow[i] * sca + rs;
#pragma unroll
      for (int t = 0; t < 4; ++t) acc[t][i] *= sca;
    }
    __syncthreads();
    // ctx += P @ V  (Vt is [d, s] so this is NT again)
#pragma unroll
    for (int kq = 0; kq < 2; ++kq) {
      const unsigned short* p0 = P + r * 64 + kq * 32 + h * 8;  // A-frag from LDS
      union { v16bf v; v8bf x[2]; } ua;
      ua.x[0] = *(const v8bf*)p0;
      ua.x[1] = *(const v8bf*)(p0 + 16);
#pragma unroll
      for (int t = 0; t < 4; ++t)
        acc[t] = wmma_bf16(ua.v, load_bfrag(vt, S_LEN, t * 16, s0 + kq * 32, lane), acc[t]);
    }
    __syncthreads();
  }
  // normalize and write ctx as bf16 [b, s, h*64+d]
  int bb = bh >> 4, head = bh & 15;
#pragma unroll
  for (int t = 0; t < 4; ++t) {
#pragma unroll
    for (int i = 0; i < 8; ++i) {
      int s = m0 + i + 8 * h;
      int n = head * HD + t * 16 + r;
      ctx[((long)bb * S_LEN + s) * DM + n] = bits2bf(f2bf_bits(acc[t][i] / lrow[i]));
    }
  }
}

extern "C" void kernel_launch(void* const* d_in, const int* in_sizes, int n_in,
                              void* d_out, int out_size, void* d_ws, size_t ws_size,
                              hipStream_t stream) {
  const float* x      = (const float*)d_in[0];
  const float* Wqkv_w = (const float*)d_in[1];
  const float* Wqkv_b = (const float*)d_in[2];
  const float* Wout_w = (const float*)d_in[3];
  const float* Wout_b = (const float*)d_in[4];
  float* out = (float*)d_out;

  char* ws = (char*)d_ws;
  const size_t MB8 = (size_t)8 * 1024 * 1024;
  bf16_t* qb    = (bf16_t*)(ws);                 // [2,16,2048,64] bf16   8 MB
  bf16_t* kb    = (bf16_t*)(ws + MB8);           // [2,16,2048,64] bf16   8 MB
  bf16_t* vtb   = (bf16_t*)(ws + 2 * MB8);       // [2,16,64,2048] bf16   8 MB
  bf16_t* ctx   = (bf16_t*)(ws + 3 * MB8);       // [4096,1024]    bf16   8 MB
  bf16_t* xb    = (bf16_t*)(ws + 4 * MB8);       // [4096,1024]    bf16   8 MB
  bf16_t* wqkvb = (bf16_t*)(ws + 5 * MB8);       // [3072,1024]    bf16   6 MB
  bf16_t* woutb = (bf16_t*)(ws + 5 * MB8 + (size_t)3072 * 1024 * 2);  // [1024,1024] 2 MB

  const int M = 2 * S_LEN;  // 4096 tokens
  // one-pass fp32 -> bf16 down-convert of activations and weights
  cvt_bf16_kernel<<<(M * DM) / (8 * 256), 256, 0, stream>>>(x,      xb,    M * DM);
  cvt_bf16_kernel<<<(3 * DM * DM) / (8 * 256), 256, 0, stream>>>(Wqkv_w, wqkvb, 3 * DM * DM);
  cvt_bf16_kernel<<<(DM * DM) / (8 * 256), 256, 0, stream>>>(Wout_w, woutb, DM * DM);
  // QKV projection + scatter (Q scaled, V transposed)
  gemm_nt_kernel<0><<<dim3(3 * DM / 128, M / 128), 256, 0, stream>>>(
      xb, wqkvb, Wqkv_b, M, 3 * DM, DM, qb, kb, vtb, nullptr);
  // flash attention
  attn_kernel<<<dim3(S_LEN / 128, 2 * NH), 256, 0, stream>>>(qb, kb, vtb, ctx);
  // output projection -> fp32
  gemm_nt_kernel<1><<<dim3(DM / 128, M / 128), 256, 0, stream>>>(
      ctx, woutb, Wout_b, M, DM, DM, nullptr, nullptr, nullptr, out);
}
